// TransformerAttention_7438883357099
// MI455X (gfx1250) — matrix operand order repaired and checked by reading, not compiled, not run on hardware
//
#include <hip/hip_runtime.h>
#include <hip/hip_bf16.h>
#include <math.h>

typedef __attribute__((ext_vector_type(16))) __bf16 v16bf;
typedef __attribute__((ext_vector_type(8)))  float  v8f;

#define N_CTX   1000
#define N_PAD   1024
#define D_MODEL 128
#define D_CAT   384
#define NHEADS  8
#define HDIM    16
#define TILE_N  32

// 16-bit A-matrix 16x32 VGPR layout (ISA 7.12.2):
// lanes 0-15 hold K = {0..7, 16..23}; lanes 16-31 hold K = {8..15, 24..31}
__device__ __forceinline__ int a_kmap(int i, int half) {
  return ((i < 8) ? i : i + 8) + 8 * half;
}

// Stage a 32-row x 128-col f32 context tile into LDS using the CDNA5
// async global->LDS path (GLOBAL_LOAD_ASYNC_TO_LDS_B128, ASYNCcnt).
// LDS dest address = low 32 bits of the flat shared pointer (ISA 10.2:
// flat->LDS mapping truncates to addr[31:0]). Rows past N_CTX are
// zero-filled with a plain DS store (async lanes are EXEC-predicated).
__device__ __forceinline__ void stage_ctx(const float* __restrict__ ctxb,
                                          int n0, float* s_ctx, int tid) {
  for (int idx = tid; idx < (TILE_N * D_MODEL) / 4; idx += 256) {
    int row = idx >> 5;           // 32 float4 per 128-float row
    int c4  = idx & 31;
    int n = n0 + row;
    float* dstp = s_ctx + row * D_MODEL + c4 * 4;
    if (n < N_CTX) {
      unsigned ldsdst = (unsigned)(unsigned long long)(void*)dstp;
      unsigned long long gaddr =
          (unsigned long long)(const void*)(ctxb + (size_t)n * D_MODEL + c4 * 4);
      asm volatile("global_load_async_to_lds_b128 %0, %1, off"
                   :: "v"(ldsdst), "v"(gaddr)
                   : "memory");
    } else {
      reinterpret_cast<float4*>(dstp)[0] = make_float4(0.f, 0.f, 0.f, 0.f);
    }
    // prefetch next tile's rows into the cache hierarchy
    int pn = n0 + TILE_N + row;
    if ((idx & 31) == 0 && pn < N_CTX)
      __builtin_prefetch(ctxb + (size_t)pn * D_MODEL, 0, 1);
  }
  // drain async copies before the workgroup barrier publishes the tile
  asm volatile("s_wait_asynccnt 0x0" ::: "memory");
}

__global__ __launch_bounds__(256)
void attn_fused_kernel(const float* __restrict__ state_t,
                       const float* __restrict__ context,
                       const unsigned char* __restrict__ mask,
                       const float* __restrict__ Wq,
                       const float* __restrict__ Wk,
                       const float* __restrict__ Wv,
                       const float* __restrict__ Wfc,
                       float* __restrict__ out) {
  __shared__ __align__(16) float s_ctx[TILE_N * D_MODEL];   // 16 KB
  __shared__ float s_scores[NHEADS * N_PAD];                // 32 KB
  __shared__ float s_wqe[NHEADS * D_MODEL];                 // 4 KB
  __shared__ float s_state[D_CAT];
  __shared__ float s_q[D_MODEL];
  __shared__ float s_ctxv[NHEADS * D_MODEL];                // 4 KB
  __shared__ float s_out[D_MODEL];

  const int bid  = blockIdx.x;
  const int tid  = threadIdx.x;
  const int lane = tid & 31;          // wave32
  const int wave = tid >> 5;          // 8 waves
  const int lm   = lane & 15;
  const int half = lane >> 4;

  const float* ctxb = context + (size_t)bid * N_CTX * D_MODEL;
  const unsigned char* maskb = mask + (size_t)bid * N_CTX;

  // ---- q = Wq @ state (tiny GEMV) ----
  for (int j = tid; j < D_CAT; j += 256)
    s_state[j] = state_t[(size_t)bid * D_CAT + j];
  __syncthreads();

  if (tid < D_MODEL) {
    float acc = 0.f;
    const float* wq = Wq + (size_t)tid * D_CAT;
    for (int j = 0; j < D_CAT; ++j) acc += s_state[j] * wq[j];
    s_q[tid] = acc;
  }
  __syncthreads();

  // ---- fold q into Wk: wq_eff[h][d] = (1/hd) * sum_j q[16h+j] * Wk[16h+j][d]
  for (int idx = tid; idx < NHEADS * D_MODEL; idx += 256) {
    int h = idx >> 7, d = idx & 127;
    float acc = 0.f;
    for (int j = 0; j < HDIM; ++j)
      acc += s_q[h * HDIM + j] * Wk[(size_t)(h * HDIM + j) * D_MODEL + d];
    s_wqe[idx] = acc * (1.f / 16.f);    // scale = 1/hd
  }
  __syncthreads();

  // ---- A registers for the scores GEMM (16x128 bf16, rows >=8 zero) ----
  v16bf aw[4];
  for (int kk = 0; kk < 4; ++kk) {
    v16bf a;
#pragma unroll
    for (int i = 0; i < 16; ++i) {
      int K = 32 * kk + a_kmap(i, half);
      float v = (lm < NHEADS) ? s_wqe[lm * D_MODEL + K] : 0.f;
      a[i] = (__bf16)v;
    }
    aw[kk] = a;
  }

  // ---- Pass 1: scores[8][N] = wq_eff x context^T, masked ----
  for (int n0 = 0; n0 < N_PAD; n0 += TILE_N) {
    stage_ctx(ctxb, n0, s_ctx, tid);
    __syncthreads();
    if (wave < 2) {                     // wave-uniform branch: EXEC all-ones inside
      const int st = wave;              // 16-column sub-tile
      v8f c = {};
      for (int kk = 0; kk < 4; ++kk) {
        v16bf bv;
        const float* src = s_ctx + (16 * st + lm) * D_MODEL + 32 * kk;
#pragma unroll
        for (int i = 0; i < 16; ++i) bv[i] = (__bf16)src[a_kmap(i, half)];
        c = __builtin_amdgcn_wmma_f32_16x16x32_bf16(false, aw[kk], false, bv,
                                                    (short)0, c, false, false);
      }
      if (half == 0) {                  // C rows 0..7 live in lanes 0-15
        int n = n0 + 16 * st + lm;
        bool msk = (n < N_CTX) ? (maskb[n] != 0) : true;
#pragma unroll
        for (int r = 0; r < NHEADS; ++r)
          s_scores[r * N_PAD + n0 + 16 * st + lm] =
              msk ? -__builtin_inff() : c[r];
      }
    }
    __syncthreads();
  }

  // ---- Softmax: wave w handles head w (exact, two reductions) ----
  {
    float* row = s_scores + wave * N_PAD;
    float mx = -__builtin_inff();
    for (int j = lane; j < N_PAD; j += 32) mx = fmaxf(mx, row[j]);
    for (int off = 16; off > 0; off >>= 1) mx = fmaxf(mx, __shfl_xor(mx, off, 32));
    float sum = 0.f;
    for (int j = lane; j < N_PAD; j += 32) {
      float e = __expf(row[j] - mx);
      row[j] = e;
      sum += e;
    }
    for (int off = 16; off > 0; off >>= 1) sum += __shfl_xor(sum, off, 32);
    float rn = 1.f / sum;
    for (int j = lane; j < N_PAD; j += 32) row[j] *= rn;
  }
  __syncthreads();

  // ---- Pass 2: ctxv[8][128] = attn x context; wave w owns d-chunk w ----
  v8f acc = {};
  for (int kt = 0; kt < N_PAD / TILE_N; ++kt) {
    stage_ctx(ctxb, kt * TILE_N, s_ctx, tid);
    __syncthreads();
    v16bf av, bv;
#pragma unroll
    for (int i = 0; i < 16; ++i) {      // A = attn tile (16x32, rows>=8 zero)
      int K = a_kmap(i, half);
      float v = (lm < NHEADS) ? s_scores[lm * N_PAD + kt * TILE_N + K] : 0.f;
      av[i] = (__bf16)v;
    }
#pragma unroll
    for (int i = 0; i < 16; ++i)        // B = context tile (32 x 16 d-cols)
      bv[i] = (__bf16)s_ctx[a_kmap(i, half) * D_MODEL + wave * HDIM + lm];
    acc = __builtin_amdgcn_wmma_f32_16x16x32_bf16(false, av, false, bv,
                                                  (short)0, acc, false, false);
    __syncthreads();
  }
  if (half == 0) {
#pragma unroll
    for (int r = 0; r < NHEADS; ++r)
      s_ctxv[r * D_MODEL + wave * HDIM + lm] = acc[r];
  }
  __syncthreads();

  // ---- out[i] = Wv_row_i . ctxv[head(i)], then final fc with Wfc ----
  if (tid < D_MODEL) {
    int h = tid >> 4;
    float a2 = 0.f;
    const float* wv = Wv + (size_t)tid * D_MODEL;
    const float* cv = s_ctxv + h * D_MODEL;
    for (int d = 0; d < D_MODEL; ++d) a2 += wv[d] * cv[d];
    s_out[tid] = a2;
  }
  __syncthreads();
  if (tid < D_MODEL) {
    float a2 = 0.f;
    const float* wf = Wfc + (size_t)tid * D_MODEL;
    for (int j = 0; j < D_MODEL; ++j) a2 += wf[j] * s_out[j];
    out[(size_t)bid * D_MODEL + tid] = a2;
  }
}

extern "C" void kernel_launch(void* const* d_in, const int* in_sizes, int n_in,
                              void* d_out, int out_size, void* d_ws, size_t ws_size,
                              hipStream_t stream) {
  const float*         state_t = (const float*)d_in[0];
  const float*         context = (const float*)d_in[1];
  const unsigned char* mask    = (const unsigned char*)d_in[2];
  const float*         Wq      = (const float*)d_in[3];
  const float*         Wk      = (const float*)d_in[4];
  const float*         Wv      = (const float*)d_in[5];
  const float*         Wfc     = (const float*)d_in[6];
  float*               outp    = (float*)d_out;

  attn_fused_kernel<<<512, 256, 0, stream>>>(state_t, context, mask,
                                             Wq, Wk, Wv, Wfc, outp);
}
